// ConvMultiScaleBlock_56607668961289
// MI455X (gfx1250) — matrix operand order repaired and checked by reading, not compiled, not run on hardware
//
#include <hip/hip_runtime.h>
#include <math.h>

// ---------------------------------------------------------------------------
// Fused windowed-attention transformer block for MI455X (gfx1250, wave32).
// Kernel 0: one-shot f32->f16 weight conversion into d_ws (stays L2-resident).
// Kernel 1: per-window  BN1 -> QKV -> attn -> proj -> +shortcut   (all in LDS)
// Kernel 2: per-64-pixel tile  BN2 -> MLP1 -> GELU -> MLP2 -> +residual
// All GEMMs via v_wmma_f32_16x16x32_f16 (f16 in, f32 accum), register-blocked
// over the token (N) dimension so each A (weight) fragment feeds 2-4 WMMAs.
// Job-derived tile indices go through readfirstlane so routing/addressing is
// scalar (s_cbranch + saddr-form loads, no EXEC churn).
// ---------------------------------------------------------------------------

typedef _Float16 f16;
typedef f16   v16h __attribute__((ext_vector_type(16)));
typedef f16   v8h  __attribute__((ext_vector_type(8)));
typedef float v8f  __attribute__((ext_vector_type(8)));
typedef float v4f  __attribute__((ext_vector_type(4)));

#define C_IN  192
#define NHEAD 3
#define HD    64
#define TOK   64       // 8x8 window tokens
#define HWSZ  16384    // 128*128
#define HID   768
#define LDC   200      // f16 stride for [tok][192ch] tiles (400B rows, 16B aligned)
#define LDV   72       // f16 stride for Vs [192ch][64tok]
#define LDH1  776      // f16 stride for H1 [tok][768]
#define EPSV  1e-5f

#define RFL(v) __builtin_amdgcn_readfirstlane(v)

// f16 weight offsets inside d_ws (elements)
#define QKVW_OFF 0
#define PW_OFF   110592            // 576*192
#define W1_OFF   147456            // + 192*192
#define W2_OFF   294912            // + 768*192
#define WTOT     442368            // + 192*768

__device__ __forceinline__ v8f wmma32(v16h a, v16h b, v8f c) {
  return __builtin_amdgcn_wmma_f32_16x16x32_f16(
      false, a, false, b, (short)0, c, false, false);
}

// ---- fragment loaders (ISA 7.12.2 wave32 layouts) ------------------------
// A 16x32: row = lane%16; elem j -> K = kb + j + (j>=8?8:0), kb = lane[4]?8:0
// B 32x16: col = lane%16; elem j -> K = kb + j + (j>=8?8:0), kb = lane[4]?8:0
// C/D:     col = lane%16; vgpr r -> M = r + (lane[4]?8:0)

__device__ __forceinline__ v16h load_a_h(const f16* W, int row, int ld, int k0, int lane) {
  const f16* p = W + (long)row * ld + k0 + ((lane & 16) ? 8 : 0);
  v8h lo = *(const v8h*)(p);
  v8h hi = *(const v8h*)(p + 16);
  v16h a;
#pragma unroll
  for (int j = 0; j < 8; ++j) { a[j] = lo[j]; a[j + 8] = hi[j]; }
  return a;
}

__device__ __forceinline__ v16h load_b_h(const f16* M, int col, int ld, int k0, int lane) {
  const f16* p = M + col * ld + k0 + ((lane & 16) ? 8 : 0);
  v8h lo = *(const v8h*)(p);
  v8h hi = *(const v8h*)(p + 16);
  v16h b;
#pragma unroll
  for (int j = 0; j < 8; ++j) { b[j] = lo[j]; b[j + 8] = hi[j]; }
  return b;
}

// bias-add (+ optional scale) then pack 8 f32 accum lanes -> v8h
__device__ __forceinline__ v8h pack8(v8f acc, v4f bl, v4f bh, float s) {
  v8h o;
#pragma unroll
  for (int r = 0; r < 4; ++r) {
    o[r]     = (f16)((acc[r]     + bl[r]) * s);
    o[r + 4] = (f16)((acc[r + 4] + bh[r]) * s);
  }
  return o;
}

// ---------------------------------------------------------------------------
// Kernel 0: f32 -> f16 weight conversion into workspace (one flat pass).
// ---------------------------------------------------------------------------
__global__ void __launch_bounds__(256)
cvt_weights_kernel(const float* __restrict__ qkvw, const float* __restrict__ pw,
                   const float* __restrict__ w1, const float* __restrict__ w2,
                   f16* __restrict__ wh) {
  int i = blockIdx.x * 256 + threadIdx.x;
  if (i >= WTOT) return;
  float v;
  if (i < PW_OFF)      v = qkvw[i];
  else if (i < W1_OFF) v = pw[i - PW_OFF];
  else if (i < W2_OFF) v = w1[i - W1_OFF];
  else                 v = w2[i - W2_OFF];
  wh[i] = (f16)v;
}

// ---------------------------------------------------------------------------
// Kernel 1: one block (256 thr = 8 waves) per 8x8 window.
// LDS map (bytes):
//   Xs [64tok][LDC] f16      @ 0       (25600)   reused for O
//   Qs [64tok][LDC] f16      @ 25600   (25600)   reused for f16 probs P
//   Ks [64tok][LDC] f16      @ 51200   (25600)
//   Vs [192ch][LDV] f16      @ 76800   (27648)
//   Sf [3][64][64] f32       @ 104448  (49152)   doubles as fp32 x staging
//   bnsc/bnsh [192] f32 each @ 153600  (1536)    -> total 155136
// ---------------------------------------------------------------------------
__global__ void __launch_bounds__(256)
win_attn_kernel(const float* __restrict__ x,
                const float* __restrict__ g1, const float* __restrict__ be1,
                const float* __restrict__ mu1, const float* __restrict__ va1,
                const f16* __restrict__ qkvwh, const float* __restrict__ qkvb,
                const f16* __restrict__ pwh,  const float* __restrict__ pb,
                float* __restrict__ x2) {
  extern __shared__ char smem[];
  f16*   Xs   = (f16*)(smem);
  f16*   Qs   = (f16*)(smem + 25600);
  f16*   Ks   = (f16*)(smem + 51200);
  f16*   Vs   = (f16*)(smem + 76800);
  float* Sf   = (float*)(smem + 104448);
  float* bnsc = (float*)(smem + 153600);
  float* bnsh = bnsc + C_IN;
  const unsigned STAGE_OFF = 104448u;   // async dest; dyn-LDS assumed at base 0

  const int tid  = threadIdx.x;
  const int wave = tid >> 5;
  const int lane = tid & 31;
  const int wid  = blockIdx.x;          // b*256 + wy*16 + wx
  const int bb   = wid >> 8;
  const int wy   = (wid >> 4) & 15;
  const int wx   = wid & 15;
  const long base = (long)bb * C_IN * HWSZ;
  const int h0 = wy * 8, w0 = wx * 8;

  // -- Phase 0: async-stage raw fp32 window (192ch x 64tok) into Sf region --
  for (int i = 0; i < 48; ++i) {
    int idx = tid + i * 256;            // idx = c*64 + t
    int c = idx >> 6, t = idx & 63;
    int h = h0 + (t >> 3), w = w0 + (t & 7);
    unsigned long long gp =
        (unsigned long long)(x + base + (long)c * HWSZ + h * 128 + w);
    unsigned lo = STAGE_OFF + (unsigned)idx * 4u;
    asm volatile("global_load_async_to_lds_b32 %0, %1, off"
                 :: "v"(lo), "v"(gp) : "memory");
  }
  // overlap: BN1 coefficients while the async DMA is in flight
  if (tid < C_IN) {
    float sc = g1[tid] * rsqrtf(va1[tid] + EPSV);
    bnsc[tid] = sc;
    bnsh[tid] = be1[tid] - mu1[tid] * sc;
  }
  asm volatile("s_wait_asynccnt 0" ::: "memory");
  __syncthreads();

  // -- Phase 1: BN1, staged fp32 [c][t] -> Xs f16 token-major [t][c] --
  {
    const float* Sx = Sf;
    int t = tid & 63, g = tid >> 6;
#pragma unroll
    for (int cc = 0; cc < 6; ++cc) {
      int c0 = g * 48 + cc * 8;
      v8h o;
#pragma unroll
      for (int r = 0; r < 8; ++r) {
        int c = c0 + r;
        o[r] = (f16)(Sx[c * 64 + t] * bnsc[c] + bnsh[c]);
      }
      *(v8h*)(Xs + t * LDC + c0) = o;
    }
  }
  __syncthreads();

  // -- Phase 2: QKV = W(576x192) * X, 2-wide over token tiles --
  for (int job = wave; job < 72; job += 8) {    // 36 M-tiles x 2 N-pairs
    int mt = RFL(job >> 1);                     // scalar q/k/v routing
    int np = RFL((job & 1) << 1);
    int row  = mt * 16 + (lane & 15);
    int col0 = np * 16 + (lane & 15);
    int col1 = col0 + 16;
    v8f acc0 = {}, acc1 = {};
#pragma unroll
    for (int kk = 0; kk < 6; ++kk) {
      v16h a  = load_a_h(qkvwh, row, C_IN, kk * 32, lane);
      v16h b0 = load_b_h(Xs, col0, LDC, kk * 32, lane);
      v16h b1 = load_b_h(Xs, col1, LDC, kk * 32, lane);
      acc0 = wmma32(a, b0, acc0);
      acc1 = wmma32(a, b1, acc1);
    }
    int orow0 = mt * 16 + ((lane & 16) ? 8 : 0);
    v4f bl = *(const v4f*)(qkvb + orow0);
    v4f bh = *(const v4f*)(qkvb + orow0 + 4);
    if (mt < 12) {                              // ---- Q (scaled) ----
      *(v8h*)(Qs + col0 * LDC + orow0) = pack8(acc0, bl, bh, 0.125f);
      *(v8h*)(Qs + col1 * LDC + orow0) = pack8(acc1, bl, bh, 0.125f);
    } else if (mt < 24) {                       // ---- K ----
      int r0 = orow0 - 192;
      *(v8h*)(Ks + col0 * LDC + r0) = pack8(acc0, bl, bh, 1.0f);
      *(v8h*)(Ks + col1 * LDC + r0) = pack8(acc1, bl, bh, 1.0f);
    } else {                                    // ---- V (channel-major) ----
      int r0 = orow0 - 384;
#pragma unroll
      for (int r = 0; r < 4; ++r) {
        Vs[(r0 + r) * LDV + col0]     = (f16)(acc0[r] + bl[r]);
        Vs[(r0 + r + 4) * LDV + col0] = (f16)(acc0[r + 4] + bh[r]);
        Vs[(r0 + r) * LDV + col1]     = (f16)(acc1[r] + bl[r]);
        Vs[(r0 + r + 4) * LDV + col1] = (f16)(acc1[r + 4] + bh[r]);
      }
    }
  }
  __syncthreads();

  // -- Phase 3: S[h][n][m] = q^T k, 2-wide over m tiles --
  for (int job = wave; job < 24; job += 8) {    // 3 heads x 4 n-tiles x 2 m-pairs
    int head = RFL(job >> 3);
    int rem  = job & 7;
    int mt = RFL(rem >> 1), np = RFL((rem & 1) << 1);
    int nrow = mt * 16 + (lane & 15);
    int m0c  = np * 16 + (lane & 15);
    v8f acc0 = {}, acc1 = {};
#pragma unroll
    for (int kk = 0; kk < 2; ++kk) {
      v16h a  = load_a_h(Qs, nrow, LDC, head * HD + kk * 32, lane);
      v16h b0 = load_b_h(Ks, m0c,      LDC, head * HD + kk * 32, lane);
      v16h b1 = load_b_h(Ks, m0c + 16, LDC, head * HD + kk * 32, lane);
      acc0 = wmma32(a, b0, acc0);
      acc1 = wmma32(a, b1, acc1);
    }
    float* Sh = Sf + head * 4096;
    int n0 = mt * 16 + ((lane & 16) ? 8 : 0);
#pragma unroll
    for (int r = 0; r < 8; ++r) {
      Sh[(n0 + r) * 64 + m0c]      = acc0[r];
      Sh[(n0 + r) * 64 + m0c + 16] = acc1[r];
    }
  }
  __syncthreads();

  // -- Phase 4: softmax; write f16 probs into retired Qs buffer --
  {
    f16* Pf = Qs;                       // [head*4096 + n*64 + m] f16
    if (tid < NHEAD * TOK) {
      float* rowp = Sf + tid * 64;
      f16*   prow = Pf + tid * 64;
      float mx = -1e30f;
      for (int m = 0; m < 64; ++m) mx = fmaxf(mx, rowp[m]);
      float s = 0.f;
      for (int m = 0; m < 64; ++m) { float e = __expf(rowp[m] - mx); rowp[m] = e; s += e; }
      float inv = 1.f / s;
      for (int m = 0; m < 64; ++m) prow[m] = (f16)(rowp[m] * inv);
    }
  }
  __syncthreads();

  // -- Phase 5: O[d][n] = sum_m V[d][m] P[n][m], 2-wide over n tiles --
  for (int job = wave; job < 24; job += 8) {
    int head = RFL(job >> 3);
    int rem  = job & 7;
    int mt = RFL(rem >> 1), np = RFL((rem & 1) << 1);
    int n0c = np * 16 + (lane & 15);
    const f16* Pf = Qs + head * 4096;
    v8f acc0 = {}, acc1 = {};
#pragma unroll
    for (int kk = 0; kk < 2; ++kk) {
      v16h a  = load_a_h(Vs, head * HD + mt * 16 + (lane & 15), LDV, kk * 32, lane);
      v16h b0 = load_b_h(Pf, n0c,      64, kk * 32, lane);
      v16h b1 = load_b_h(Pf, n0c + 16, 64, kk * 32, lane);
      acc0 = wmma32(a, b0, acc0);
      acc1 = wmma32(a, b1, acc1);
    }
    int ch0 = head * HD + mt * 16 + ((lane & 16) ? 8 : 0);
    v8h o0, o1;
#pragma unroll
    for (int r = 0; r < 8; ++r) { o0[r] = (f16)acc0[r]; o1[r] = (f16)acc1[r]; }
    *(v8h*)(Xs + n0c * LDC + ch0) = o0;
    *(v8h*)(Xs + (n0c + 16) * LDC + ch0) = o1;
  }
  __syncthreads();

  // -- Phase 6: proj(192x192) * O + bias + shortcut -> x2, 2-wide --
  for (int job = wave; job < 24; job += 8) {    // 12 M-tiles x 2 N-pairs
    int mt = RFL(job >> 1), np = RFL((job & 1) << 1);
    int row  = mt * 16 + (lane & 15);
    int col0 = np * 16 + (lane & 15);
    int col1 = col0 + 16;
    v8f acc0 = {}, acc1 = {};
#pragma unroll
    for (int kk = 0; kk < 6; ++kk) {
      v16h a  = load_a_h(pwh, row, C_IN, kk * 32, lane);
      v16h b0 = load_b_h(Xs, col0, LDC, kk * 32, lane);
      v16h b1 = load_b_h(Xs, col1, LDC, kk * 32, lane);
      acc0 = wmma32(a, b0, acc0);
      acc1 = wmma32(a, b1, acc1);
    }
    int r0 = mt * 16 + ((lane & 16) ? 8 : 0);
    v4f bl = *(const v4f*)(pb + r0);
    v4f bh = *(const v4f*)(pb + r0 + 4);
    int ha = h0 + (col0 >> 3), wa = w0 + (col0 & 7);
    int hb = h0 + (col1 >> 3), wb = w0 + (col1 & 7);
#pragma unroll
    for (int r = 0; r < 8; ++r) {
      int ch = r0 + r;
      float bias = (r < 4) ? bl[r] : bh[r - 4];
      long gia = base + (long)ch * HWSZ + ha * 128 + wa;
      long gib = base + (long)ch * HWSZ + hb * 128 + wb;
      x2[gia] = acc0[r] + bias + x[gia];
      x2[gib] = acc1[r] + bias + x[gib];
    }
  }
}

// ---------------------------------------------------------------------------
// Kernel 2: one block per 64 contiguous pixels. BN2 -> W1+GELU (hidden stays
// in LDS) -> W2 + residual. x2in aliases out (read-before-write per element).
// LDS: XB [64][LDC] f16 @0 (25600) | H1 [64][LDH1] f16 @25600 (99328)
//      bnsc/bnsh [192] f32 @124928 (1536)  -> total 126464
// ---------------------------------------------------------------------------
__global__ void __launch_bounds__(256)
mlp_kernel(const float* x2in,
           const float* __restrict__ g2, const float* __restrict__ be2,
           const float* __restrict__ mu2, const float* __restrict__ va2,
           const f16* __restrict__ w1h, const float* __restrict__ b1,
           const f16* __restrict__ w2h, const float* __restrict__ b2,
           float* out) {
  extern __shared__ char smem[];
  f16*   XB   = (f16*)(smem);
  f16*   H1   = (f16*)(smem + 25600);
  float* bnsc = (float*)(smem + 124928);
  float* bnsh = bnsc + C_IN;

  const int tid = threadIdx.x, wave = tid >> 5, lane = tid & 31;
  const int p0  = blockIdx.x * 64;
  const int bbt = p0 >> 14;
  const int hw  = p0 & 16383;
  const long base = (long)bbt * C_IN * HWSZ + hw;

  if (tid < C_IN) {
    float sc = g2[tid] * rsqrtf(va2[tid] + EPSV);
    bnsc[tid] = sc;
    bnsh[tid] = be2[tid] - mu2[tid] * sc;
  }
  __syncthreads();

  // BN2 into XB token-major [t][c]
  {
    int t = tid & 63, g = tid >> 6;
#pragma unroll
    for (int cc = 0; cc < 6; ++cc) {
      int c0 = g * 48 + cc * 8;
      v8h o;
#pragma unroll
      for (int r = 0; r < 8; ++r) {
        int c = c0 + r;
        o[r] = (f16)(x2in[base + (long)c * HWSZ + t] * bnsc[c] + bnsh[c]);
      }
      *(v8h*)(XB + t * LDC + c0) = o;
    }
  }
  __syncthreads();

  // H1 = gelu(W1(768x192) * XB + b1), 4-wide over token tiles (48 jobs = 6/wave)
  for (int job = wave; job < 48; job += 8) {
    int mt = RFL(job);
    int row = mt * 16 + (lane & 15);
    int colb = lane & 15;
    v8f acc0 = {}, acc1 = {}, acc2 = {}, acc3 = {};
#pragma unroll
    for (int kk = 0; kk < 6; ++kk) {
      v16h a   = load_a_h(w1h, row, C_IN, kk * 32, lane);
      v16h fb0 = load_b_h(XB, colb,      LDC, kk * 32, lane);
      v16h fb1 = load_b_h(XB, colb + 16, LDC, kk * 32, lane);
      v16h fb2 = load_b_h(XB, colb + 32, LDC, kk * 32, lane);
      v16h fb3 = load_b_h(XB, colb + 48, LDC, kk * 32, lane);
      acc0 = wmma32(a, fb0, acc0);
      acc1 = wmma32(a, fb1, acc1);
      acc2 = wmma32(a, fb2, acc2);
      acc3 = wmma32(a, fb3, acc3);
    }
    int r0 = mt * 16 + ((lane & 16) ? 8 : 0);
    v4f bl = *(const v4f*)(b1 + r0);
    v4f bh = *(const v4f*)(b1 + r0 + 4);
#pragma unroll
    for (int nt = 0; nt < 4; ++nt) {
      v8f acc = (nt == 0) ? acc0 : (nt == 1) ? acc1 : (nt == 2) ? acc2 : acc3;
      v8h o;
#pragma unroll
      for (int r = 0; r < 8; ++r) {
        float h = acc[r] + ((r < 4) ? bl[r] : bh[r - 4]);
        h = 0.5f * h * (1.f + erff(h * 0.70710678118f));   // exact GELU
        o[r] = (f16)h;
      }
      *(v8h*)(H1 + (colb + nt * 16) * LDH1 + r0) = o;
    }
  }
  __syncthreads();

  // out = W2(192x768) * H1 + b2 + x2, 2-wide (24 jobs = 3/wave)
  for (int job = wave; job < 24; job += 8) {
    int mt = RFL(job >> 1), np = RFL((job & 1) << 1);
    int row  = mt * 16 + (lane & 15);
    int col0 = np * 16 + (lane & 15);
    int col1 = col0 + 16;
    v8f acc0 = {}, acc1 = {};
    for (int kk = 0; kk < 24; ++kk) {
      v16h a   = load_a_h(w2h, row, HID, kk * 32, lane);
      v16h fb0 = load_b_h(H1, col0, LDH1, kk * 32, lane);
      v16h fb1 = load_b_h(H1, col1, LDH1, kk * 32, lane);
      acc0 = wmma32(a, fb0, acc0);
      acc1 = wmma32(a, fb1, acc1);
    }
    int r0 = mt * 16 + ((lane & 16) ? 8 : 0);
    v4f bl = *(const v4f*)(b2 + r0);
    v4f bh = *(const v4f*)(b2 + r0 + 4);
#pragma unroll
    for (int r = 0; r < 8; ++r) {
      int ch = r0 + r;
      float bias = (r < 4) ? bl[r] : bh[r - 4];
      long gia = base + (long)ch * HWSZ + col0;
      long gib = base + (long)ch * HWSZ + col1;
      out[gia] = acc0[r] + bias + x2in[gia];
      out[gib] = acc1[r] + bias + x2in[gib];
    }
  }
}

// ---------------------------------------------------------------------------
extern "C" void kernel_launch(void* const* d_in, const int* in_sizes, int n_in,
                              void* d_out, int out_size, void* d_ws, size_t ws_size,
                              hipStream_t stream) {
  (void)in_sizes; (void)n_in; (void)out_size; (void)ws_size;
  const float* x    = (const float*)d_in[0];
  const float* g1   = (const float*)d_in[1];
  const float* be1  = (const float*)d_in[2];
  const float* mu1  = (const float*)d_in[3];
  const float* va1  = (const float*)d_in[4];
  const float* qkvw = (const float*)d_in[5];
  const float* qkvb = (const float*)d_in[6];
  const float* pw   = (const float*)d_in[7];
  const float* pb   = (const float*)d_in[8];
  const float* g2   = (const float*)d_in[9];
  const float* be2  = (const float*)d_in[10];
  const float* mu2  = (const float*)d_in[11];
  const float* va2  = (const float*)d_in[12];
  const float* w1   = (const float*)d_in[13];
  const float* b1   = (const float*)d_in[14];
  const float* w2   = (const float*)d_in[15];
  const float* b2   = (const float*)d_in[16];
  float* out = (float*)d_out;
  f16*   wh  = (f16*)d_ws;             // f16 weights live here

  // Kernel 0: convert weights once per call (L2 keeps them hot afterwards).
  cvt_weights_kernel<<<dim3((WTOT + 255) / 256), dim3(256), 0, stream>>>(
      qkvw, pw, w1, w2, wh);

  const size_t shm1 = 155136;   // Xs+Qs+Ks+Vs+Sf+bn coef
  const size_t shm2 = 126464;   // XB+H1+bn coef

  // x2 = shortcut + attn lives in d_out between the two kernels.
  win_attn_kernel<<<dim3(4096), dim3(256), shm1, stream>>>(
      x, g1, be1, mu1, va1, wh + QKVW_OFF, qkvb, wh + PW_OFF, pb, out);
  mlp_kernel<<<dim3(4096), dim3(256), shm2, stream>>>(
      out, g2, be2, mu2, va2, wh + W1_OFF, b1, wh + W2_OFF, b2, out);
}
